// ResBlock_50019189129549
// MI455X (gfx1250) — matrix operand order repaired and checked by reading, not compiled, not run on hardware
//
#include <hip/hip_runtime.h>

// ---------------------------------------------------------------------------
// CDNA5 (gfx1250) fused sparse-conv residual block.
//   spconv = gather(+fused lrelu/BN) -> per-offset GEMM (v_wmma bf16) ->
//            scatter-add (global_atomic_add_f32).
// Weights are pre-converted once to transposed bf16 [K][64][Cin] and staged
// per-block into LDS via the Tensor Data Mover (tensor_load_to_lds), using the
// TDM pad feature to create the bank-conflict-free padded LDS stride.
// All feature maps (25.6 MB) are L2-resident on the 192 MB L2, so the WMMA
// path exists to keep the 44 GFLOP of GEMM off the VALU critical path.
// ---------------------------------------------------------------------------

typedef __attribute__((ext_vector_type(16))) __bf16 v16bf;
typedef __attribute__((ext_vector_type(8)))  __bf16 bf16x8;
typedef __attribute__((ext_vector_type(4)))  __bf16 bf16x4;
typedef __attribute__((ext_vector_type(8)))  float  v8f;
typedef __attribute__((ext_vector_type(4)))  float  f32x4;
typedef __attribute__((ext_vector_type(4)))  unsigned int u32x4;
typedef __attribute__((ext_vector_type(8)))  int i32x8;
typedef __attribute__((ext_vector_type(4)))  int i32x4;

#define LRELU(v) ((v) > 0.0f ? (v) : 0.01f * (v))

// ---------------------------------------------------------------------------
// TDM: DMA one [rows=64][cin] bf16 tile (contiguous in global) into LDS,
// inserting 16 B of padding after every row so LDS stride = cin+8 halves.
// Descriptor bit layout per CDNA5 ISA §8.3 (group 0) / §8.4 (group 1).
// ---------------------------------------------------------------------------
__device__ __forceinline__ void tdm_load_w_tile(const __bf16* gsrc,
                                                void* ldsDst, int cin) {
  const unsigned long long ga = (unsigned long long)(uintptr_t)gsrc;
  u32x4 g0;
  g0[0] = 1u;                                      // count=1, user mode
  g0[1] = (unsigned int)(uintptr_t)ldsDst;         // lds_addr (bytes)
  g0[2] = (unsigned int)(ga & 0xFFFFFFFFu);        // global_addr[31:0]
  g0[3] = (unsigned int)((ga >> 32) & 0x01FFFFFFu) // global_addr[56:32]
          | 0x80000000u;                           // type=2 ("image")
  const unsigned int rows = 64u;
  const unsigned int padI = (cin == 64) ? 4u : 3u; // row = 32/16 dwords
  const unsigned int padA = 3u;                    // +4 dwords (16B) per row
  const unsigned long long st1 = (unsigned long long)cin * rows;
  i32x8 g1;
  g1[0] = (int)((1u << 16)            // data_size = 2 bytes
                | (1u << 20)          // pad_enable
                | (padI << 22) | (padA << 25));
  g1[1] = (int)(((unsigned)cin & 0xFFFFu) << 16);                 // dim0 lo16
  g1[2] = (int)((((unsigned)cin >> 16) & 0xFFFFu) | (rows << 16));// dim0hi|dim1lo
  g1[3] = (int)(((rows >> 16) & 0xFFFFu) | (((unsigned)cin & 0xFFFFu) << 16)); // dim1hi|tile0
  g1[4] = (int)(rows & 0xFFFFu);                                  // tile1, tile2=0
  g1[5] = (int)(unsigned)cin;                                     // dim0_stride[31:0]
  g1[6] = (int)(((unsigned)(st1 & 0xFFFFu)) << 16);               // dim1_stride lo16
  g1[7] = (int)(unsigned)(st1 >> 16);                             // dim1_stride hi
  const i32x4 z4 = {0, 0, 0, 0};
#if __clang_major__ >= 23
  const i32x8 z8 = {0, 0, 0, 0, 0, 0, 0, 0};
  __builtin_amdgcn_tensor_load_to_lds(g0, g1, z4, z4, z8, 0);
#else
  __builtin_amdgcn_tensor_load_to_lds(g0, g1, z4, z4, 0);
#endif
}

// ---------------------------------------------------------------------------
__global__ void zero_kernel(float* __restrict__ p, long long n) {
  long long i = (long long)blockIdx.x * blockDim.x + threadIdx.x;
  const long long stride = (long long)gridDim.x * blockDim.x;
  for (; i < n; i += stride) p[i] = 0.0f;
}

// Convert W [K][Cin][64] fp32 -> Wt [K][64][Cin] bf16 (transposed, TDM-ready).
__global__ void convert_weights_kernel(const float* __restrict__ W,
                                       __bf16* __restrict__ out, int K,
                                       int Cin) {
  const int total = K * Cin * 64;
  const int i = blockIdx.x * 256 + threadIdx.x;
  if (i >= total) return;
  const int c = i % Cin;
  const int t = i / Cin;
  const int d = t & 63;
  const int k = t >> 6;
  out[i] = (__bf16)W[((long long)k * Cin + c) * 64 + d];
}

// ---------------------------------------------------------------------------
// spconv: block = (64-row tile = blockIdx.x, offset k = blockIdx.y),
// 128 threads = 4 waves; wave w computes rows [16w,16w+16) x 64 cols.
// ---------------------------------------------------------------------------
template <int CIN_T>
__global__ __launch_bounds__(128) void spconv_wmma_kernel(
    const float* __restrict__ feats, const __bf16* __restrict__ Wt,
    const int* __restrict__ in_idx, const int* __restrict__ out_idx,
    const float* __restrict__ sb, float* __restrict__ out, int P) {
  constexpr int ROWS = 64;
  constexpr int AP = CIN_T + 8;  // padded strides -> conflict-free b128 reads
  constexpr int BP = CIN_T + 8;  // matches TDM pad (16B per row)
  __shared__ __align__(16) __bf16 lA[ROWS * AP];  // A tile, row-major [r][c]
  __shared__ __align__(16) __bf16 lB[64 * BP];    // W[k] transposed [col][c]

  const int k = blockIdx.y;
  const int rowBase = blockIdx.x * ROWS;
  const int tid = threadIdx.x;
  const long long kOff = (long long)k * P;

  // --- async: DMA W[k] (bf16, pre-transposed) into LDS via TDM (wave 0) ---
  if (tid < 32) tdm_load_w_tile(Wt + (long long)k * (64 * CIN_T), (void*)lB, CIN_T);

  // --- gather + transform A tile (ROWS x CIN) into LDS (bf16) ---
  constexpr int C4 = CIN_T / 4;
  for (int idx = tid; idx < ROWS * C4; idx += 128) {
    const int r = idx / C4;
    const int c4 = (idx - r * C4) * 4;
    const int p = rowBase + r;
    f32x4 v = {0.0f, 0.0f, 0.0f, 0.0f};
    if (p < P) {
      const int src = in_idx[kOff + p];
      v = *(const f32x4*)(feats + (long long)src * CIN_T + c4);
      if (sb) {
#pragma unroll
        for (int j = 0; j < 4; ++j) {
          float t = v[j];
          t = LRELU(t);
          v[j] = t * sb[c4 + j] + sb[64 + c4 + j];
        }
      }
    }
    bf16x4 h;
#pragma unroll
    for (int j = 0; j < 4; ++j) h[j] = (__bf16)v[j];
    *(bf16x4*)(&lA[r * AP + c4]) = h;
  }
  if (tid < 32) __builtin_amdgcn_s_wait_tensorcnt(0);
  __syncthreads();

  const int lane = tid & 31;
  const int wave = tid >> 5;
  const int m = lane & 15;
  const int aKb = (lane < 16) ? 0 : 8;   // A 16x32 lane layout (ISA 7.12.2)
  const int bKb = (lane < 16) ? 0 : 8;   // B 32x16 lane layout

  v8f acc[4] = {v8f{}, v8f{}, v8f{}, v8f{}};

#pragma unroll
  for (int kc = 0; kc < CIN_T / 32; ++kc) {
    const int k0 = kc * 32;
    // A fragment: lanes 0-15 K {0..7,16..23}, lanes 16-31 K {8..15,24..31}
    const __bf16* Ar = &lA[(wave * 16 + m) * AP + k0 + aKb];
    const bf16x8 alo = *(const bf16x8*)(Ar);
    const bf16x8 ahi = *(const bf16x8*)(Ar + 16);
    const v16bf a = __builtin_shufflevector(alo, ahi, 0, 1, 2, 3, 4, 5, 6, 7,
                                            8, 9, 10, 11, 12, 13, 14, 15);
#pragma unroll
    for (int ct = 0; ct < 4; ++ct) {
      // B fragment: lB is [col][K]; lanes 0-15 K {0..7,16..23}, lanes 16-31 K {8..15,24..31}
      const __bf16* Br = &lB[(ct * 16 + m) * BP + k0 + bKb];
      const bf16x8 blo = *(const bf16x8*)(Br);
      const bf16x8 bhi = *(const bf16x8*)(Br + 16);
      const v16bf b = __builtin_shufflevector(blo, bhi, 0, 1, 2, 3, 4, 5, 6, 7,
                                              8, 9, 10, 11, 12, 13, 14, 15);
      acc[ct] = __builtin_amdgcn_wmma_f32_16x16x32_bf16(
          false, a, false, b, (short)0, acc[ct], false, false);
    }
  }

  // --- scatter-add D (16x64 per wave) via rulebook out indices ---
#pragma unroll
  for (int ct = 0; ct < 4; ++ct) {
    const int col = ct * 16 + m;
#pragma unroll
    for (int r = 0; r < 8; ++r) {
      const int mRow = (lane < 16) ? r : (8 + r);  // C/D VGPR layout
      const int p = rowBase + wave * 16 + mRow;
      if (p < P) {
        const int o = out_idx[kOff + p];
        unsafeAtomicAdd(&out[(long long)o * 64 + col], acc[ct][r]);
      }
    }
  }
}

// ---------------------------------------------------------------------------
// Per-channel sum / sum-sq of lrelu(raw) -> stats[0:64], stats[64:128].
// ---------------------------------------------------------------------------
__global__ __launch_bounds__(256) void stats_kernel(
    const float* __restrict__ raw, int n, float* __restrict__ stats) {
  __shared__ float ssum[256];
  __shared__ float ssq[256];
  const int c = threadIdx.x & 63;
  const int g = threadIdx.x >> 6;
  const int rowBase = blockIdx.x * 64 + g * 16;
  float s = 0.0f, q = 0.0f;
#pragma unroll
  for (int i = 0; i < 16; ++i) {
    const int r = rowBase + i;
    if (r < n) {
      float v = raw[(long long)r * 64 + c];
      v = LRELU(v);
      s += v;
      q += v * v;
    }
  }
  ssum[threadIdx.x] = s;
  ssq[threadIdx.x] = q;
  __syncthreads();
  if (g == 0) {
    s = ssum[c] + ssum[64 + c] + ssum[128 + c] + ssum[192 + c];
    q = ssq[c] + ssq[64 + c] + ssq[128 + c] + ssq[192 + c];
    unsafeAtomicAdd(&stats[c], s);
    unsafeAtomicAdd(&stats[64 + c], q);
  }
}

// stats -> fused BN scale/bias:  y = x*scale + bias
__global__ void bn_finalize_kernel(const float* __restrict__ stats,
                                   const float* __restrict__ gamma,
                                   const float* __restrict__ beta, float invN,
                                   float* __restrict__ sb) {
  const int c = threadIdx.x;  // 64 threads
  const float mean = stats[c] * invN;
  const float var = stats[64 + c] * invN - mean * mean;
  const float s = gamma[c] * rsqrtf(var + 1e-5f);
  sb[c] = s;
  sb[64 + c] = beta[c] - mean * s;
}

// resA = bn2(lrelu(conv3_raw)) + bn02(lrelu(conv12_raw))
__global__ void combine_kernel(const float* __restrict__ r2raw,
                               const float* __restrict__ s2raw,
                               const float* __restrict__ sb2,
                               const float* __restrict__ sb02,
                               float* __restrict__ resA, long long total) {
  const long long i = (long long)blockIdx.x * 256 + threadIdx.x;
  if (i >= total) return;
  const int c = (int)(i & 63);
  float a = r2raw[i];
  a = LRELU(a);
  a = a * sb2[c] + sb2[64 + c];
  float b = s2raw[i];
  b = LRELU(b);
  b = b * sb02[c] + sb02[64 + c];
  resA[i] = a + b;
}

// ---------------------------------------------------------------------------
static void launch_spconv(int cin, const float* feats, const __bf16* Wt,
                          const int* in_idx, const int* out_idx,
                          const float* sb, float* out, int P, int K,
                          hipStream_t stream) {
  const int tiles = (P + 63) / 64;
  const dim3 grid((unsigned)tiles, (unsigned)K);
  if (cin == 32)
    spconv_wmma_kernel<32><<<grid, 128, 0, stream>>>(feats, Wt, in_idx,
                                                     out_idx, sb, out, P);
  else
    spconv_wmma_kernel<64><<<grid, 128, 0, stream>>>(feats, Wt, in_idx,
                                                     out_idx, sb, out, P);
}

extern "C" void kernel_launch(void* const* d_in, const int* in_sizes, int n_in,
                              void* d_out, int out_size, void* d_ws,
                              size_t ws_size, hipStream_t stream) {
  const float* x   = (const float*)d_in[0];
  const float* W1  = (const float*)d_in[1];
  const float* W12 = (const float*)d_in[2];
  const float* W2  = (const float*)d_in[3];
  const float* W3  = (const float*)d_in[4];
  const float* Wp  = (const float*)d_in[5];
  const float* g0  = (const float*)d_in[6];
  const float* b0  = (const float*)d_in[7];
  const float* g02 = (const float*)d_in[8];
  const float* b02 = (const float*)d_in[9];
  const float* g1  = (const float*)d_in[10];
  const float* b1  = (const float*)d_in[11];
  const float* g2  = (const float*)d_in[12];
  const float* b2  = (const float*)d_in[13];
  const int* rbAi = (const int*)d_in[14];
  const int* rbAo = (const int*)d_in[15];
  const int* rbBi = (const int*)d_in[16];
  const int* rbBo = (const int*)d_in[17];
  const int* rbPi = (const int*)d_in[18];
  const int* rbPo = (const int*)d_in[19];

  const int n = in_sizes[0] / 32;            // N active voxels
  const int m = out_size / 64 - n;           // M pooled voxels
  const float invN = 1.0f / (float)n;

  float* outF = (float*)d_out;
  float* resB = outF;                        // [M,64]
  float* resA = outF + (long long)m * 64;    // [N,64]

  const size_t NC = (size_t)n * 64;
  float* wsf  = (float*)d_ws;
  float* bufA = wsf;                 // conv1 raw, later reused for conv2 raw
  float* bufB = wsf + NC;            // conv12 raw
  float* bufC = wsf + 2 * NC;        // conv3 raw
  float* stats = wsf + 3 * NC;       // 128 floats, reused per layer
  float* sb0  = stats + 128;
  float* sb02 = sb0 + 128;
  float* sb1  = sb02 + 128;
  float* sb2  = sb1 + 128;
  // bf16 transposed weights (TDM source), 16B-aligned
  __bf16* W1t  = (__bf16*)(sb2 + 128);
  __bf16* W12t = W1t + 9 * 64 * 32;
  __bf16* W2t  = W12t + 9 * 64 * 64;
  __bf16* W3t  = W2t + 9 * 64 * 32;
  __bf16* Wpt  = W3t + 9 * 64 * 64;

  auto zero = [&](float* p, long long cnt) {
    long long blocks = (cnt + 255) / 256;
    if (blocks > 4096) blocks = 4096;
    zero_kernel<<<dim3((unsigned)blocks), 256, 0, stream>>>(p, cnt);
  };
  auto convW = [&](const float* W, __bf16* out, int K, int Cin) {
    const int total = K * Cin * 64;
    convert_weights_kernel<<<dim3((unsigned)((total + 255) / 256)), 256, 0,
                             stream>>>(W, out, K, Cin);
  };
  const int statBlocks = (n + 63) / 64;

  // --- weight conversion (tiny; runs once per call, stream-ordered) ---
  convW(W1, W1t, 9, 32);
  convW(W12, W12t, 9, 64);
  convW(W2, W2t, 9, 32);
  convW(W3, W3t, 9, 64);
  convW(Wp, Wpt, 27, 64);

  // --- shortcut branch: conv1 (3x1) -> bn0 -> conv1_2 (1x3) -> bn02 ---
  zero(bufA, (long long)NC);
  launch_spconv(32, x, W1t, rbAi, rbAo, nullptr, bufA, n, 9, stream);
  zero(stats, 128);
  stats_kernel<<<statBlocks, 256, 0, stream>>>(bufA, n, stats);
  bn_finalize_kernel<<<1, 64, 0, stream>>>(stats, g0, b0, invN, sb0);

  zero(bufB, (long long)NC);
  launch_spconv(64, bufA, W12t, rbBi, rbBo, sb0, bufB, n, 9, stream);
  zero(stats, 128);
  stats_kernel<<<statBlocks, 256, 0, stream>>>(bufB, n, stats);
  bn_finalize_kernel<<<1, 64, 0, stream>>>(stats, g02, b02, invN, sb02);

  // --- main branch: conv2 (1x3) -> bn1 -> conv3 (3x1) -> bn2 ---
  zero(bufA, (long long)NC);  // reuse (conv12 already consumed it)
  launch_spconv(32, x, W2t, rbBi, rbBo, nullptr, bufA, n, 9, stream);
  zero(stats, 128);
  stats_kernel<<<statBlocks, 256, 0, stream>>>(bufA, n, stats);
  bn_finalize_kernel<<<1, 64, 0, stream>>>(stats, g1, b1, invN, sb1);

  zero(bufC, (long long)NC);
  launch_spconv(64, bufA, W3t, rbAi, rbAo, sb1, bufC, n, 9, stream);
  zero(stats, 128);
  stats_kernel<<<statBlocks, 256, 0, stream>>>(bufC, n, stats);
  bn_finalize_kernel<<<1, 64, 0, stream>>>(stats, g2, b2, invN, sb2);

  // --- residual add into resA (lives in d_out, also pool conv's input) ---
  combine_kernel<<<dim3((unsigned)((NC + 255) / 256)), 256, 0, stream>>>(
      bufC, bufB, sb2, sb02, resA, (long long)NC);

  // --- strided pooling conv: resA -> resB ---
  zero(resB, (long long)m * 64);
  launch_spconv(64, resA, Wpt, rbPi, rbPo, nullptr, resB, n, 27, stream);
}
